// KMeansVectorQuantizer_58686433133152
// MI455X (gfx1250) — matrix operand order repaired and checked by reading, not compiled, not run on hardware
//
#include <hip/hip_runtime.h>
#include <hip/hip_bf16.h>
#include <math.h>

// VQ quantizer for MI455X (gfx1250, wave32).
// Workspace layout (requires ~102 MB of d_ws):
//   x_f32   : 67108864 B  @ 0
//   x_bf16  : 33554432 B  @ 67108864
//   w_bf16  :   262144 B  @ 100663296
//   e_bf16  :   524288 B  @ 100925440
//   e_norm  :     4096 B  @ 101449728
//   q_idx   :   262144 B  @ 101453824
//   hist    :     4096 B  @ 101715968
//   sse     :       64 B  @ 101720064

typedef __attribute__((ext_vector_type(16))) __bf16 v16bf;
typedef __attribute__((ext_vector_type(8)))  float  v8f;
typedef __attribute__((ext_vector_type(8)))  int    v8i;
typedef __attribute__((ext_vector_type(4)))  int    v4i;
typedef __attribute__((ext_vector_type(4)))  unsigned int v4u;

#if defined(__has_builtin)
#  if __has_builtin(__builtin_amdgcn_tensor_load_to_lds)
#    define VQ_HAS_TDM 1
#  endif
#endif
#ifndef VQ_HAS_TDM
#  define VQ_HAS_TDM 0
#endif

__device__ __forceinline__ unsigned short f32_to_bf16(float f) {
  unsigned int u = __builtin_bit_cast(unsigned int, f);
  unsigned int r = u + 0x7FFFu + ((u >> 16) & 1u);  // round-to-nearest-even
  return (unsigned short)(r >> 16);
}
__device__ __forceinline__ unsigned int pack_bf16x2(float lo, float hi) {
  return (unsigned int)f32_to_bf16(lo) | ((unsigned int)f32_to_bf16(hi) << 16);
}
__device__ __forceinline__ v8f wmma_bf16(v8i a, v8i b, v8f c) {
  return __builtin_amdgcn_wmma_f32_16x16x32_bf16(
      false, __builtin_bit_cast(v16bf, a),
      false, __builtin_bit_cast(v16bf, b),
      (short)0, c, false, false);
}

__device__ __forceinline__ void wait_tensorcnt_0() {
#if defined(__has_builtin) && __has_builtin(__builtin_amdgcn_s_wait_tensorcnt)
  __builtin_amdgcn_s_wait_tensorcnt(0);
#else
  asm volatile("s_wait_tensorcnt 0x0" ::: "memory");
#endif
}
__device__ __forceinline__ void wait_tensorcnt_1() {
#if defined(__has_builtin) && __has_builtin(__builtin_amdgcn_s_wait_tensorcnt)
  __builtin_amdgcn_s_wait_tensorcnt(1);
#else
  asm volatile("s_wait_tensorcnt 0x1" ::: "memory");
#endif
}

#if VQ_HAS_TDM
// TDM: DMA nrows x 256 bf16 row-major tile (row stride 256) into LDS, padding
// +4 dwords every 128 dwords -> LDS row pitch 132 dwords (bank-conflict-free).
// D# per CDNA5 ISA: group0 = {count|flags, lds_addr,
// global_addr lo, global_addr hi | type=2}; group1 packs data_size/pad/dims/strides.
__device__ __forceinline__ void tdm_load_rows_padded(
    unsigned int lds_byte, unsigned long long gaddr, unsigned int tensor_rows) {
  v4u g0;
  g0[0] = 1u;                                    // count=1, user mode
  g0[1] = lds_byte;                              // lds_addr
  g0[2] = (unsigned int)gaddr;                   // global_addr[31:0]
  g0[3] = (unsigned int)(gaddr >> 32) | (2u << 30);  // global_addr[56:32] | type=2
  v8i g1;
  g1[0] = (int)((1u << 16)      // data_size = 1 (2 bytes)
              | (1u << 20)      // pad_enable
              | (6u << 22)      // pad_interval: every 128 dwords
              | (3u << 25));    // pad_amount: 4 dwords
  g1[1] = (int)((256u & 0xFFFFu) << 16);               // tensor_dim0[15:0]=256
  g1[2] = (int)((tensor_rows & 0xFFFFu) << 16);        // dim0 hi=0 | tensor_dim1 lo
  g1[3] = (int)((tensor_rows >> 16) | (256u << 16));   // tensor_dim1 hi | tile_dim0=256
  g1[4] = 64;                                          // tile_dim1=64, tile_dim2=0
  g1[5] = 256;                                         // tensor_dim0_stride=256
  g1[6] = 0;                                           // stride0 hi | stride1 lo
  g1[7] = 0;                                           // stride1 hi
  v4i z4 = {0, 0, 0, 0};
#if defined(__clang_major__) && __clang_major__ >= 23
  v8i z8 = {0, 0, 0, 0, 0, 0, 0, 0};
  __builtin_amdgcn_tensor_load_to_lds(g0, g1, z4, z4, z8, 0);
#else
  __builtin_amdgcn_tensor_load_to_lds(g0, g1, z4, z4, 0);
#endif
}
#endif  // VQ_HAS_TDM

// ---------------------------------------------------------------- prep
__global__ __launch_bounds__(256) void vq_prep(
    const float* __restrict__ proj_w, const float* __restrict__ embed,
    unsigned short* __restrict__ w_bf, unsigned short* __restrict__ e_bf,
    float* __restrict__ e_norm)
{
  const int bid = blockIdx.x, tid = threadIdx.x;
  if (bid < 512) {                       // proj_w [256,512] -> bf16
    const int i = bid * 256 + tid;
    w_bf[i] = f32_to_bf16(proj_w[i]);
  } else if (bid < 1536) {               // embed [1024,256] -> bf16
    const int i = (bid - 512) * 256 + tid;
    e_bf[i] = f32_to_bf16(embed[i]);
  } else {                               // ||e||^2 per code (f32)
    const int row = bid - 1536;
    const float x = embed[row * 256 + tid];
    float s = x * x;
    #pragma unroll
    for (int m = 16; m >= 1; m >>= 1) s += __shfl_xor(s, m);
    __shared__ float red[8];
    if ((tid & 31) == 0) red[tid >> 5] = s;
    __syncthreads();
    if (tid == 0) {
      float t = 0.f;
      #pragma unroll
      for (int i = 0; i < 8; i++) t += red[i];
      e_norm[row] = t;
    }
  }
}

// ------------------------------------------------------- GEMM1: projection
// x[m=(b,t), e] = sum_c inputs[b,c,t] * w[e,c] + bias[e]
// block: 512 thr = 16 waves (4x4), 64 rows(t) x 256 cols(e) per block.
__global__ __launch_bounds__(512) void vq_proj_gemm(
    const float* __restrict__ inputs, const unsigned short* __restrict__ w_bf,
    const float* __restrict__ proj_b, float* __restrict__ x_f32,
    unsigned short* __restrict__ x_bf)
{
  __shared__ float        lds_a[32 * 64];    // A tile f32, [c][t]
  __shared__ unsigned int lds_b[256 * 17];   // B tile bf16 pairs, padded pitch 17 dw

  const int tid = threadIdx.x;
  const int m0  = blockIdx.x * 64;
  const int bb  = m0 >> 12;          // batch
  const int t0l = m0 & 4095;         // t offset in batch
  const int wave = tid >> 5, lane = tid & 31;
  const int wr = wave >> 2, wc = wave & 3;
  const int ln = lane & 15, hi = (lane >> 4) & 1;

  v8f acc[4];
  v8f zero = {0.f,0.f,0.f,0.f,0.f,0.f,0.f,0.f};
  #pragma unroll
  for (int i = 0; i < 4; i++) acc[i] = zero;

  const int ac  = tid >> 4;           // 0..31 (c within tile)
  const int atq = (tid & 15) << 2;    // 0..60 (t within tile)
  const int be  = tid >> 1;           // 0..255 (e row)
  const int bh  = tid & 1;

  for (int ks = 0; ks < 16; ++ks) {
    const int k0 = ks * 32;
    __syncthreads();
    { // A: inputs[bb][k0+ac][t0l+atq..+3], coalesced float4
      const float4 v = *(const float4*)(inputs +
          ((size_t)(bb * 512 + k0 + ac)) * 4096 + t0l + atq);
      *(float4*)(lds_a + ac * 64 + atq) = v;
    }
    { // B: w_bf row be, 32 bf16 (16 dwords), two uint4 per thread-half
      const uint4* bp = (const uint4*)w_bf;
      const uint4 q0 = bp[(size_t)be * 64 + (k0 >> 3) + bh * 2];
      const uint4 q1 = bp[(size_t)be * 64 + (k0 >> 3) + bh * 2 + 1];
      unsigned int* d = lds_b + be * 17 + bh * 8;
      d[0]=q0.x; d[1]=q0.y; d[2]=q0.z; d[3]=q0.w;
      d[4]=q1.x; d[5]=q1.y; d[6]=q1.z; d[7]=q1.w;
    }
    __syncthreads();

    const int t = wr * 16 + ln;
    v8i ai;
    #pragma unroll
    for (int v = 0; v < 8; v++) {   // A-frag: lane<16 K{0..7,16..23}, lane>=16 +8
      const int kb = ((v < 4) ? 2 * v : 2 * v + 8) + hi * 8;
      ai[v] = (int)pack_bf16x2(lds_a[kb * 64 + t], lds_a[(kb + 1) * 64 + t]);
    }
    #pragma unroll
    for (int nt = 0; nt < 4; nt++) {
      const int n = wc * 64 + nt * 16 + ln;
      v8i bi;
      #pragma unroll
      for (int v = 0; v < 8; v++) bi[v] = (int)lds_b[n * 17 + ((v < 4) ? v : v + 4) + hi * 4];
      acc[nt] = wmma_bf16(ai, bi, acc[nt]);
    }
  }
  // epilogue: +bias, store f32 + bf16 (C layout: row = v + 8*hi, col = lane%16)
  #pragma unroll
  for (int nt = 0; nt < 4; nt++) {
    const int n = wc * 64 + nt * 16 + ln;
    const float bias = proj_b[n];
    #pragma unroll
    for (int v = 0; v < 8; v++) {
      const size_t m = (size_t)(m0 + wr * 16 + v + 8 * hi);
      const float val = acc[nt][v] + bias;
      x_f32[m * 256 + n] = val;
      x_bf [m * 256 + n] = f32_to_bf16(val);
    }
  }
}

// ------------------------------------------- GEMM2: distances + argmin
// score = ||e||^2 - 2 x.e ; argmin over 1024 codes.
// block: 128 thr = 4 waves, 64 rows per block; A-frags held in 64 VGPRs.
// TDM path: double-buffered tensor_load_to_lds (pad feature makes pitch-132 rows)
// overlapped with the 32 WMMAs per chunk; wave 0 drives the DMA + tensorcnt.
__device__ __forceinline__ void vq_chunk_compute(
    const unsigned int* __restrict__ sb, const v8i* a_i, int cbase, int ln, int hi,
    const float* __restrict__ e_norm, float* mval, int* midx)
{
  #pragma unroll
  for (int tile = 0; tile < 4; ++tile) {
    v8f c = {0.f,0.f,0.f,0.f,0.f,0.f,0.f,0.f};
    const int n = tile * 16 + ln;
    #pragma unroll
    for (int s = 0; s < 8; s++) {
      v8i bi;
      #pragma unroll
      for (int v = 0; v < 8; v++) bi[v] = (int)sb[n * 132 + ((v < 4) ? v : v + 4) + hi * 4 + 16 * s];
      c = wmma_bf16(a_i[s], bi, c);
    }
    const int code = cbase + n;
    const float en = e_norm[code];
    #pragma unroll
    for (int v = 0; v < 8; v++) {
      const float sc = fmaf(-2.0f, c[v], en);
      if (sc < mval[v]) { mval[v] = sc; midx[v] = code; }  // codes increase: keeps first
    }
  }
}

__global__ __launch_bounds__(128) void vq_dist_argmin(
    const unsigned short* __restrict__ x_bf, const unsigned short* __restrict__ e_bf,
    const float* __restrict__ e_norm, int* __restrict__ q_idx)
{
#if VQ_HAS_TDM
  __shared__ unsigned int smem[2 * 8448];   // two pitch-132 buffers (2 x 33792 B)
#else
  __shared__ unsigned int smem[8448];
#endif
  const int tid = threadIdx.x;
  const int m0 = blockIdx.x * 64;
  const int wave = tid >> 5, lane = tid & 31;
  const int ln = lane & 15, hi = (lane >> 4) & 1;
  const int r = tid >> 1, half = tid & 1;
  (void)r; (void)half;

#if VQ_HAS_TDM
  const unsigned int lds_base = (unsigned int)(size_t)(const void*)smem;
  const unsigned long long e_base = (unsigned long long)(size_t)e_bf;
  if (wave == 0) {   // DMA 64 x-rows into buffer 0 (HW applies pitch-132 padding)
    tdm_load_rows_padded(lds_base,
        (unsigned long long)(size_t)x_bf + (unsigned long long)m0 * 512ull, 65536u);
    wait_tensorcnt_0();
  }
  __syncthreads();
#else
  { // stage 64 x-rows (bf16, 128 dw each) into padded LDS manually
    const uint4* xp = (const uint4*)x_bf;
    uint4* dp = (uint4*)smem;
    #pragma unroll
    for (int j = 0; j < 16; j++)
      dp[r * 33 + half * 16 + j] = xp[(size_t)(m0 + r) * 32 + half * 16 + j];
  }
  __syncthreads();
#endif

  const int trow = wave * 16 + ln;
  v8i a_i[8];                       // 8 k-steps of A in registers
  #pragma unroll
  for (int s = 0; s < 8; s++) {
    #pragma unroll
    for (int v = 0; v < 8; v++) {
      const int dv = ((v < 4) ? v : v + 4) + hi * 4 + 16 * s;
      a_i[s][v] = (int)smem[trow * 132 + dv];
    }
  }

  float mval[8]; int midx[8];
  #pragma unroll
  for (int v = 0; v < 8; v++) { mval[v] = 3.4e38f; midx[v] = 0; }

#if VQ_HAS_TDM
  __syncthreads();                  // everyone done reading x before buf0 reuse
  if (wave == 0) tdm_load_rows_padded(lds_base, e_base, 1024u);  // chunk 0 -> buf 0
  for (int chunk = 0; chunk < 16; ++chunk) {
    const unsigned int* sb = smem + (chunk & 1) * 8448;
    if (wave == 0) {
      if (chunk + 1 < 16) {        // prefetch next chunk into the other buffer
        tdm_load_rows_padded(lds_base + (unsigned)((chunk + 1) & 1) * 33792u,
                             e_base + (unsigned long long)(chunk + 1) * 32768ull,
                             1024u);
        wait_tensorcnt_1();        // current chunk's DMA done (in-order)
      } else {
        wait_tensorcnt_0();
      }
    }
    __syncthreads();
    vq_chunk_compute(sb, a_i, chunk * 64, ln, hi, e_norm, mval, midx);
    __syncthreads();               // done reading sb before it is refilled
  }
#else
  for (int chunk = 0; chunk < 16; ++chunk) {
    const int cbase = chunk * 64;
    __syncthreads();
    { // stage 64 codebook rows (bf16)
      const uint4* ep = (const uint4*)e_bf;
      uint4* dp = (uint4*)smem;
      #pragma unroll
      for (int j = 0; j < 16; j++)
        dp[r * 33 + half * 16 + j] = ep[(size_t)(cbase + r) * 32 + half * 16 + j];
    }
    __syncthreads();
    vq_chunk_compute(smem, a_i, cbase, ln, hi, e_norm, mval, midx);
  }
#endif

  // reduce across the 16 lanes holding one row (first-index tie-break)
  #pragma unroll
  for (int v = 0; v < 8; v++) {
    float bv = mval[v]; int bi_ = midx[v];
    #pragma unroll
    for (int mask = 1; mask < 16; mask <<= 1) {
      const float ov = __shfl_xor(bv, mask);
      const int   oi = __shfl_xor(bi_, mask);
      if (ov < bv || (ov == bv && oi < bi_)) { bv = ov; bi_ = oi; }
    }
    if (ln == 0) q_idx[m0 + wave * 16 + v + 8 * hi] = bi_;
  }
}

// ---------------------------------- gather + transpose-out + loss + histogram
// block: 256 thr (one e per thread), 32 t per block.
__global__ __launch_bounds__(256) void vq_gather_out(
    const int* __restrict__ q_idx, const float* __restrict__ embed,
    const float* __restrict__ x_f32, float* __restrict__ out,
    unsigned int* __restrict__ hist, float* __restrict__ sse)
{
  const int tid = threadIdx.x;
  const int t0  = blockIdx.x * 32;
  const int bb  = t0 >> 12;
  const int t0l = t0 & 4095;

  if (tid < 32) atomicAdd(&hist[q_idx[t0 + tid]], 1u);

  float z[32]; float ls = 0.0f;
  #pragma unroll
  for (int tt = 0; tt < 32; ++tt) {
    const int idx = q_idx[t0 + tt];                    // uniform -> scalar load
    const float val = embed[(size_t)idx * 256 + tid];  // coalesced row read
    z[tt] = val;
    const float xv = x_f32[(size_t)(t0 + tt) * 256 + tid];
    const float d = val - xv;
    ls = fmaf(d, d, ls);
  }
  float4* op = (float4*)(out + ((size_t)bb * 256 + tid) * 4096 + t0l);
  #pragma unroll
  for (int j = 0; j < 8; j++) {
    float4 q; q.x = z[4*j]; q.y = z[4*j+1]; q.z = z[4*j+2]; q.w = z[4*j+3];
    op[j] = q;
  }
  #pragma unroll
  for (int m = 16; m >= 1; m >>= 1) ls += __shfl_xor(ls, m);
  __shared__ float red[8];
  if ((tid & 31) == 0) red[tid >> 5] = ls;
  __syncthreads();
  if (tid == 0) {
    float s = 0.f;
    #pragma unroll
    for (int i = 0; i < 8; i++) s += red[i];
    atomicAdd(sse, s);
  }
}

// ----------------------------------------------------------- finalize scalars
__global__ __launch_bounds__(1024) void vq_finalize(
    const unsigned int* __restrict__ hist, const float* __restrict__ sse,
    float* __restrict__ out)
{
  const int tid = threadIdx.x;
  const float p = (float)hist[tid] * (1.0f / 65536.0f);
  float v = -p * logf(p + 1e-10f);
  #pragma unroll
  for (int m = 16; m >= 1; m >>= 1) v += __shfl_xor(v, m);
  __shared__ float red[32];
  if ((tid & 31) == 0) red[tid >> 5] = v;
  __syncthreads();
  if (tid == 0) {
    float H = 0.f;
    #pragma unroll
    for (int i = 0; i < 32; i++) H += red[i];
    out[16777216] = 1.25f * sse[0] * (1.0f / 16777216.0f);  // (1+0.25)*mean
    out[16777233] = H;                                      // log_perplexity
  }
  if (tid < 16) out[16777217 + tid] = logf(1024.0f) * 4096.0f;  // kldiv_r
}

extern "C" void kernel_launch(void* const* d_in, const int* in_sizes, int n_in,
                              void* d_out, int out_size, void* d_ws, size_t ws_size,
                              hipStream_t stream) {
  const float* inputs = (const float*)d_in[0];   // [16,512,4096]
  const float* proj_w = (const float*)d_in[1];   // [256,512]
  const float* proj_b = (const float*)d_in[2];   // [256]
  const float* embed  = (const float*)d_in[3];   // [1024,256]
  float* out = (float*)d_out;
  char* ws = (char*)d_ws;

  float*          x_f32  = (float*)(ws);
  unsigned short* x_bf   = (unsigned short*)(ws + 67108864);
  unsigned short* w_bf   = (unsigned short*)(ws + 100663296);
  unsigned short* e_bf   = (unsigned short*)(ws + 100925440);
  float*          e_norm = (float*)(ws + 101449728);
  int*            q_idx  = (int*)(ws + 101453824);
  unsigned int*   hist   = (unsigned int*)(ws + 101715968);
  float*          sse    = (float*)(ws + 101720064);

  hipMemsetAsync(ws + 101715968, 0, 4096 + 64, stream);  // hist + sse

  vq_prep       <<<2560, 256, 0, stream>>>(proj_w, embed, w_bf, e_bf, e_norm);
  vq_proj_gemm  <<<1024, 512, 0, stream>>>(inputs, w_bf, proj_b, x_f32, x_bf);
  vq_dist_argmin<<<1024, 128, 0, stream>>>(x_bf, e_bf, e_norm, q_idx);
  vq_gather_out <<<2048, 256, 0, stream>>>(q_idx, embed, x_f32, out, hist, sse);
  vq_finalize   <<<1, 1024, 0, stream>>>(hist, sse, out);
}
